// NN_k_NN_regression_45028437131834
// MI455X (gfx1250) — matrix operand order repaired and checked by reading, not compiled, not run on hardware
//
#include <hip/hip_runtime.h>

typedef __attribute__((ext_vector_type(16))) _Float16 v16h;
typedef __attribute__((ext_vector_type(8)))  float    v8f;
typedef __attribute__((ext_vector_type(4)))  float    f4;
typedef unsigned int u32x4 __attribute__((ext_vector_type(4)));
typedef int          i32x4 __attribute__((ext_vector_type(4)));
typedef int          i32x8 __attribute__((ext_vector_type(8)));

#define B_Q      128
#define C_CASES  20000
#define F_FEAT   64
#define TOPK     32
#define EPSF     1e-10f
#define K2_THREADS 256

__device__ __forceinline__ float fast_sigmoid(float x) {
  // sigmoid(x) = 1 / (1 + exp(-x)); exp via v_exp_f32, recip via v_rcp_f32
  float e = __builtin_amdgcn_exp2f(-x * 1.4426950408889634f);
  return __builtin_amdgcn_rcpf(1.0f + e);
}

// Compute 8 feature activations for contiguous f0..f0+7 (all 16B-aligned rows).
struct G8 { float v[8]; };
__device__ __forceinline__ G8 sig8(const float* __restrict__ q,
                                   const float* __restrict__ cs,
                                   const float* __restrict__ wf,
                                   const float* __restrict__ bf,
                                   int f0) {
  G8 r;
  f4 qa = *(const f4*)(q  + f0);  f4 qb = *(const f4*)(q  + f0 + 4);
  f4 ca = *(const f4*)(cs + f0);  f4 cb = *(const f4*)(cs + f0 + 4);
  f4 wa = *(const f4*)(wf + f0);  f4 wb = *(const f4*)(wf + f0 + 4);
  f4 ba = *(const f4*)(bf + f0);  f4 bb = *(const f4*)(bf + f0 + 4);
#pragma unroll
  for (int j = 0; j < 4; ++j) {
    float d0 = qa[j] - ca[j];
    r.v[j]     = fast_sigmoid(ba[j] - d0 * d0 * wa[j]);
    float d1 = qb[j] - cb[j];
    r.v[4 + j] = fast_sigmoid(bb[j] - d1 * d1 * wb[j]);
  }
  return r;
}

// ---------------------------------------------------------------------------
// Kernel 1: one wave per 16-case tile. Computes fa (f32 output) directly in
// the WMMA 16-bit A-operand lane/VGPR layout, reduces over F with
// v_wmma_f32_16x16x32_f16 (B = w_ca broadcast to all 16 columns), then
// applies the case sigmoid and stores ca[b, tile*16 .. +15] to workspace.
// ---------------------------------------------------------------------------
__global__ __launch_bounds__(32)
void fa_ca_kernel(const float* __restrict__ query,   // [B,F]
                  const float* __restrict__ cases,   // [C,F]
                  const float* __restrict__ w_fa,    // [F]
                  const float* __restrict__ b_fa,    // [F]
                  const float* __restrict__ w_ca,    // [F]
                  const float* __restrict__ b_ca,    // [1]
                  float* __restrict__ fa_out,        // [B,C,F]
                  float* __restrict__ ca_ws)         // [B,C]
{
  const int tilesPerB = C_CASES / 16;           // 1250 exact
  const int b    = blockIdx.x / tilesPerB;
  const int tile = blockIdx.x % tilesPerB;
  const int lane = threadIdx.x;                 // 0..31, full wave
  const int mi   = lane & 15;                   // A-matrix row M
  const int hi   = lane >> 4;                   // K-half selector
  const int m    = tile * 16 + mi;              // case index for this lane

  const float* qrow = query + (size_t)b * F_FEAT;
  const float* crow = cases + (size_t)m * F_FEAT;
  float* farow = fa_out + ((size_t)b * C_CASES + (size_t)m) * F_FEAT;

  v8f acc = {0.f, 0.f, 0.f, 0.f, 0.f, 0.f, 0.f, 0.f};

#pragma unroll
  for (int c0 = 0; c0 < F_FEAT; c0 += 32) {
    // A-operand K mapping (16-bit A 16x32):
    //   lane<16 : elements 0..7 -> K=c0+0..7,  elements 8..15 -> K=c0+16..23
    //   lane>=16: elements 0..7 -> K=c0+8..15, elements 8..15 -> K=c0+24..31
    const int fA = c0 + hi * 8;
    const int fB = c0 + 16 + hi * 8;

    G8 gA = sig8(qrow, crow, w_fa, b_fa, fA);
    G8 gB = sig8(qrow, crow, w_fa, b_fa, fB);

    // f32 feature_activations output (each lane: two 32B-contiguous groups)
    f4 sA0 = {gA.v[0], gA.v[1], gA.v[2], gA.v[3]};
    f4 sA1 = {gA.v[4], gA.v[5], gA.v[6], gA.v[7]};
    f4 sB0 = {gB.v[0], gB.v[1], gB.v[2], gB.v[3]};
    f4 sB1 = {gB.v[4], gB.v[5], gB.v[6], gB.v[7]};
    *(f4*)(farow + fA)     = sA0;
    *(f4*)(farow + fA + 4) = sA1;
    *(f4*)(farow + fB)     = sB0;
    *(f4*)(farow + fB + 4) = sB1;

    v16h a;
#pragma unroll
    for (int j = 0; j < 8; ++j) {
      a[j]     = (_Float16)gA.v[j];
      a[8 + j] = (_Float16)gB.v[j];
    }

    // B-operand (16-bit B 32x16): same K mapping as A (elements 0..7 ->
    // K=c0+8*hi+0..7, elements 8..15 -> K=c0+16+8*hi+0..7); identical columns broadcast w_ca.
    const int wcb = c0 + hi * 8;
    f4 w0 = *(const f4*)(w_ca + wcb);
    f4 w1 = *(const f4*)(w_ca + wcb + 4);
    f4 w2 = *(const f4*)(w_ca + wcb + 16);
    f4 w3 = *(const f4*)(w_ca + wcb + 20);
    v16h bb;
#pragma unroll
    for (int j = 0; j < 4; ++j) {
      bb[j]      = (_Float16)w0[j];
      bb[4 + j]  = (_Float16)w1[j];
      bb[8 + j]  = (_Float16)w2[j];
      bb[12 + j] = (_Float16)w3[j];
    }

    acc = __builtin_amdgcn_wmma_f32_16x16x32_f16(
        /*neg_a=*/false, a, /*neg_b=*/false, bb,
        /*c_mod=*/(short)0, acc, /*reuse_a=*/false, /*reuse_b=*/false);
  }

  // D layout: lane l, vgpr r -> M = r + 8*hi (every column identical).
  const float bca = b_ca[0];
  float sca[8];
#pragma unroll
  for (int r = 0; r < 8; ++r) sca[r] = fast_sigmoid(acc[r] + bca);

  if (mi == 0) {  // lanes 0 and 16 cover cases tile*16+0..7 and +8..15
    float* dst = ca_ws + (size_t)b * C_CASES + (size_t)(tile * 16 + hi * 8);
    f4 lo = {sca[0], sca[1], sca[2], sca[3]};
    f4 hi4 = {sca[4], sca[5], sca[6], sca[7]};
    *(f4*)(dst)     = lo;
    *(f4*)(dst + 4) = hi4;
  }
}

// ---------------------------------------------------------------------------
// Kernel 2: one block per query row. TDM-stage the ca row into LDS, iterative
// top-32 argmax with in-LDS sign-masking, then normalize / scatter / reduce.
// ---------------------------------------------------------------------------
__global__ __launch_bounds__(K2_THREADS)
void topk_kernel(const float* __restrict__ ca_ws,   // [B,C]
                 const float* __restrict__ labels,  // [C]
                 float* __restrict__ ca_out,        // [B,C]
                 float* __restrict__ knn_out,       // [B]
                 float* __restrict__ pred_out)      // [B]
{
  __shared__ __align__(16) float sca[C_CASES];      // 80 KB (LDS: 320KB/WGP)
  __shared__ float rv[K2_THREADS];
  __shared__ int   ri[K2_THREADS];
  __shared__ float topv[TOPK];
  __shared__ int   topi[TOPK];
  __shared__ float sinv;

  const int b   = blockIdx.x;
  const int tid = threadIdx.x;
  const float* src = ca_ws + (size_t)b * C_CASES;

#if __has_builtin(__builtin_amdgcn_tensor_load_to_lds) && __has_builtin(__builtin_amdgcn_s_wait_tensorcnt)
  if (tid < 32) {  // wave 0 issues one TDM descriptor for the whole row
    unsigned lds_addr = (unsigned)(uintptr_t)(void*)sca;  // flat low 32b = LDS offset
    unsigned long long ga = (unsigned long long)(uintptr_t)src;
    u32x4 g0;
    g0[0] = 1u;                                    // count=1, user mode
    g0[1] = lds_addr;                              // lds_addr [63:32]
    g0[2] = (unsigned)(ga & 0xffffffffu);          // global_addr [95:64]
    g0[3] = (unsigned)((ga >> 32) & 0x01ffffffu) | (2u << 30);  // addr hi + type=2
    i32x8 g1;
    g1[0] = (int)(2u << 16);                       // data_size=4B, no flags
    g1[1] = (int)((C_CASES & 0xffffu) << 16);      // tensor_dim0 lo16
    g1[2] = (int)((C_CASES >> 16) | (1u << 16));   // tensor_dim0 hi16, tensor_dim1=1
    g1[3] = (int)((C_CASES & 0xffffu) << 16);      // tile_dim0 = C
    g1[4] = 0;                                     // tile_dim1/2 unused
    g1[5] = (int)C_CASES;                          // tensor_dim0_stride lo32
    g1[6] = 0;
    g1[7] = 0;
    i32x4 gz = {0, 0, 0, 0};
#if defined(__clang_major__) && __clang_major__ >= 23
    i32x8 gz8 = {0, 0, 0, 0, 0, 0, 0, 0};
    __builtin_amdgcn_tensor_load_to_lds(g0, g1, gz, gz, gz8, 0);
#else
    __builtin_amdgcn_tensor_load_to_lds(g0, g1, gz, gz, 0);
#endif
    __builtin_amdgcn_s_wait_tensorcnt(0);
  }
#else
  for (int i = tid * 4; i < C_CASES; i += K2_THREADS * 4) {
    *(f4*)(&sca[i]) = *(const f4*)(src + i);
  }
#endif
  __syncthreads();

  // --- iterative top-32 (strict >, ties -> lowest index, like lax.top_k) ---
  for (int k = 0; k < TOPK; ++k) {
    float bv = -1.0f;
    int   bi = 0x7fffffff;
    for (int i = tid; i < C_CASES; i += K2_THREADS) {
      float v = sca[i];
      if (v > bv) { bv = v; bi = i; }
    }
    rv[tid] = bv; ri[tid] = bi;
    __syncthreads();
    for (int s = K2_THREADS / 2; s > 0; s >>= 1) {
      if (tid < s) {
        float ov = rv[tid + s]; int oi = ri[tid + s];
        float mv = rv[tid];     int mi = ri[tid];
        if (ov > mv || (ov == mv && oi < mi)) { rv[tid] = ov; ri[tid] = oi; }
      }
      __syncthreads();
    }
    if (tid == 0) {
      topv[k] = rv[0]; topi[k] = ri[0];
      sca[ri[0]] = -rv[0];   // mask: all real values > 0.5, sign marks selection
    }
    __syncthreads();
  }

  // --- denom, kNN mean, weighted prediction ---
  if (tid == 0) {
    float denom = 0.f;
    for (int k = 0; k < TOPK; ++k) denom += topv[k];
    denom += EPSF;
    float inv = 1.0f / denom;
    float mean = 0.f, pred = 0.f;
    for (int k = 0; k < TOPK; ++k) {
      float lbl = labels[topi[k]];
      mean += lbl;
      pred += topv[k] * inv * lbl;
    }
    knn_out[b]  = mean * (1.0f / (float)TOPK);
    pred_out[b] = pred;
    sinv = inv;
  }
  __syncthreads();

  // --- normalized case_activations row (zeros except selected) ---
  const float inv = sinv;
  float* dst = ca_out + (size_t)b * C_CASES;
  for (int i = tid; i < C_CASES; i += K2_THREADS) {
    float v = sca[i];
    dst[i] = (v < 0.0f) ? (-v) * inv : 0.0f;
  }
}

extern "C" void kernel_launch(void* const* d_in, const int* in_sizes, int n_in,
                              void* d_out, int out_size, void* d_ws, size_t ws_size,
                              hipStream_t stream) {
  const float* query  = (const float*)d_in[0];
  const float* cases  = (const float*)d_in[1];
  const float* labels = (const float*)d_in[2];
  const float* w_fa   = (const float*)d_in[3];
  const float* b_fa   = (const float*)d_in[4];
  const float* w_ca   = (const float*)d_in[5];
  const float* b_ca   = (const float*)d_in[6];
  // d_in[7] = top_k (32, baked in as TOPK)

  float* out      = (float*)d_out;
  float* fa_out   = out;                                          // [B,C,F]
  float* ca_out   = out + (size_t)B_Q * C_CASES * F_FEAT;         // [B,C]
  float* knn_out  = ca_out + (size_t)B_Q * C_CASES;               // [B]
  float* pred_out = knn_out + B_Q;                                // [B]
  float* ca_ws    = (float*)d_ws;                                 // [B,C] scratch

  dim3 grid1(B_Q * (C_CASES / 16));
  fa_ca_kernel<<<grid1, 32, 0, stream>>>(query, cases, w_fa, b_fa, w_ca, b_ca,
                                         fa_out, ca_ws);
  topk_kernel<<<B_Q, K2_THREADS, 0, stream>>>(ca_ws, labels, ca_out,
                                              knn_out, pred_out);
}
